// MultiHeadAttention_23699629540037
// MI455X (gfx1250) — matrix operand order repaired and checked by reading, not compiled, not run on hardware
//
#include <hip/hip_runtime.h>

// ---------------------------------------------------------------------------
// Multi-head attention forward for MI455X (gfx1250, wave32, WMMA bf16).
// B=4, N=2048, D=1024, H=16, HD=64.  All GEMMs via v_wmma_f32_16x16x32_bf16.
// Double-buffered LDS staging in the projections; 64-key flash-attention
// blocks; block-uniform q/k/v scatter (no per-element branches).
// ---------------------------------------------------------------------------

typedef __bf16 bf16;
typedef __attribute__((ext_vector_type(16))) __bf16 v16bf;
typedef __attribute__((ext_vector_type(8)))  __bf16 v8bf;
typedef __attribute__((ext_vector_type(8)))  float  v8f;

union BFrag { v16bf v; v8bf h[2]; };

static __device__ __forceinline__ v8f wmma_bf16(v16bf a, v16bf b, v8f c) {
  // D = A(16x32 bf16) * B(32x16 bf16) + C(16x16 f32)
  return __builtin_amdgcn_wmma_f32_16x16x32_bf16(false, a, false, b,
                                                 (short)0, c, false, false);
}

#define BATCH 4
#define SEQ   2048
#define EMBED 1024
#define HEADS 16
#define HDIM  64

#define LDS_STRIDE 40   // 32 K-halves + 8 pad -> 80B rows (16B aligned)
#define PSTRIDE    72   // 64 key-halves + 8 pad -> 144B rows (16B aligned)

// ===========================================================================
// Kernel 1: qkv = x @ W_qkv^T   (M=8192, K=1024, Ncols=3072)
// 256 threads = 8 waves; block tile 128x128; wave tile 64x32 (4x2 wmma frags).
// Double-buffered LDS.  Writes Q,K bf16 [bh][n][hd]; V transposed [bh][hd][n].
// ===========================================================================
__global__ __launch_bounds__(256)
void mha_qkv_gemm(const float* __restrict__ x, const float* __restrict__ Wqkv,
                  bf16* __restrict__ q_ws, bf16* __restrict__ k_ws,
                  bf16* __restrict__ v_ws)
{
  __shared__ alignas(16) bf16 xs [2][128 * LDS_STRIDE];
  __shared__ alignas(16) bf16 wsb[2][128 * LDS_STRIDE];

  const int tid  = threadIdx.x;
  const int lane = tid & 31;
  const int wave = tid >> 5;
  const int ll   = lane & 15;        // lane within half-wave
  const int lg   = lane >> 4;        // half-wave group
  const int mbase = blockIdx.x * 128;
  const int nbase = blockIdx.y * 128;
  const int mw = (wave >> 2) * 64;   // wave row offset in tile
  const int nw = (wave &  3) * 32;   // wave col offset in tile

  const v8f vzero = {0.f,0.f,0.f,0.f,0.f,0.f,0.f,0.f};
  v8f acc[4][2];
  #pragma unroll
  for (int mt = 0; mt < 4; ++mt)
    #pragma unroll
    for (int nt = 0; nt < 2; ++nt) acc[mt][nt] = vzero;

  const int kb0 = lg ? 8  : 0;   // A-frag half-wave K offset
  const int kb2 = lg ? 16 : 0;   // unused: B-frag also uses kb0

  // stage one 128x32 fp32 tile pair -> bf16 LDS buffers
  auto stage = [&](int buf, int kb) {
    #pragma unroll
    for (int i = 0; i < 4; ++i) {
      const int idx = tid + i * 256;        // 1024 float4 units per tile
      const int row = idx >> 3;             // 8 float4 per 32-wide row
      const int c4  = (idx & 7) * 4;
      const float4 fx = *(const float4*)(x    + (size_t)(mbase + row) * EMBED + kb + c4);
      const float4 fw = *(const float4*)(Wqkv + (size_t)(nbase + row) * EMBED + kb + c4);
      bf16* dx = &xs [buf][row * LDS_STRIDE + c4];
      bf16* dw = &wsb[buf][row * LDS_STRIDE + c4];
      dx[0] = (bf16)fx.x; dx[1] = (bf16)fx.y; dx[2] = (bf16)fx.z; dx[3] = (bf16)fx.w;
      dw[0] = (bf16)fw.x; dw[1] = (bf16)fw.y; dw[2] = (bf16)fw.z; dw[3] = (bf16)fw.w;
    }
  };

  stage(0, 0);
  __syncthreads();

  for (int kb = 0; kb < EMBED; kb += 32) {
    const int buf = (kb >> 5) & 1;
    if (kb + 32 < EMBED) stage(buf ^ 1, kb + 32);   // overlap with wmma below

    BFrag a[4], b[2];
    #pragma unroll
    for (int mt = 0; mt < 4; ++mt) {
      const bf16* p = &xs[buf][(mw + mt * 16 + ll) * LDS_STRIDE];
      a[mt].h[0] = *(const v8bf*)(p + kb0);        // K 0..7 | 8..15
      a[mt].h[1] = *(const v8bf*)(p + 16 + kb0);   // K 16..23 | 24..31
    }
    #pragma unroll
    for (int nt = 0; nt < 2; ++nt) {
      const bf16* p = &wsb[buf][(nw + nt * 16 + ll) * LDS_STRIDE + kb0];
      b[nt].h[0] = *(const v8bf*)(p);
      b[nt].h[1] = *(const v8bf*)(p + 16);
    }
    #pragma unroll
    for (int mt = 0; mt < 4; ++mt)
      #pragma unroll
      for (int nt = 0; nt < 2; ++nt)
        acc[mt][nt] = wmma_bf16(a[mt].v, b[nt].v, acc[mt][nt]);
    __syncthreads();   // next iter: reads buf^1 (staged above), rewrites buf
  }

  // ---- scatter into q/k/v layouts; sel is BLOCK-uniform (128 | 1024) ----
  const int sel = nbase >> 10;                      // 0=Q 1=K 2=V
  if (sel == 2) {
    #pragma unroll
    for (int nt = 0; nt < 2; ++nt) {
      const int d  = (nbase + nw + nt * 16 + ll) & 1023;
      const int h  = d >> 6, hd = d & 63;
      #pragma unroll
      for (int mt = 0; mt < 4; ++mt)
        #pragma unroll
        for (int r = 0; r < 8; ++r) {
          const int mm = mbase + mw + mt * 16 + r + lg * 8;
          const int bi = mm >> 11, nr = mm & 2047;
          v_ws[(((size_t)(bi * HEADS + h)) * HDIM + hd) * SEQ + nr] =
              (bf16)acc[mt][nt][r];                 // transposed [bh][hd][n]
        }
    }
  } else {
    bf16* __restrict__ dst = sel ? k_ws : q_ws;
    #pragma unroll
    for (int nt = 0; nt < 2; ++nt) {
      const int d  = (nbase + nw + nt * 16 + ll) & 1023;
      const int h  = d >> 6, hd = d & 63;
      #pragma unroll
      for (int mt = 0; mt < 4; ++mt)
        #pragma unroll
        for (int r = 0; r < 8; ++r) {
          const int mm = mbase + mw + mt * 16 + r + lg * 8;
          const int bi = mm >> 11, nr = mm & 2047;
          dst[(((size_t)(bi * HEADS + h)) * SEQ + nr) * HDIM + hd] =
              (bf16)acc[mt][nt][r];
        }
    }
  }
}

// ===========================================================================
// Kernel 2: flash attention.  grid = B*H*(N/128); 8 waves, 16 q-rows per wave.
// 64-key blocks: 8 wmma (S = Q K^T), online softmax (4 shfl reductions per
// row per 64 keys), LDS transpose of P, 8 wmma (O += P V).
// ===========================================================================
__global__ __launch_bounds__(256)
void mha_flash_attn(const bf16* __restrict__ q_ws, const bf16* __restrict__ k_ws,
                    const bf16* __restrict__ v_ws, bf16* __restrict__ attn_ws)
{
  __shared__ alignas(16) bf16 p_lds[8][16 * PSTRIDE];   // per-wave P tile

  const int tid  = threadIdx.x;
  const int lane = tid & 31;
  const int wave = tid >> 5;
  const int ll   = lane & 15;
  const int lg   = lane >> 4;

  const int gx = blockIdx.x;           // B*H*16
  const int qb = gx & 15;
  const int bh = gx >> 4;              // 0..63
  const int qbase = qb * 128 + wave * 16;

  const bf16* qp = q_ws + ((size_t)bh * SEQ + qbase) * HDIM;
  const bf16* kp = k_ws + (size_t)bh * SEQ * HDIM;
  const bf16* vp = v_ws + (size_t)bh * HDIM * SEQ;

  const int kb0 = lg ? 8  : 0;
  const int kb2 = lg ? 16 : 0;

  // Q A-fragments (2 chunks of K=32 over HDIM=64), resident in VGPRs
  BFrag qf[2];
  #pragma unroll
  for (int c2 = 0; c2 < 2; ++c2) {
    const bf16* p = qp + (size_t)ll * HDIM + c2 * 32;
    qf[c2].h[0] = *(const v8bf*)(p + kb0);
    qf[c2].h[1] = *(const v8bf*)(p + 16 + kb0);
  }

  const v8f vzero = {0.f,0.f,0.f,0.f,0.f,0.f,0.f,0.f};
  float m[8], l[8];
  v8f o[4];
  #pragma unroll
  for (int r = 0; r < 8; ++r) { m[r] = -1e30f; l[r] = 0.f; }
  #pragma unroll
  for (int f = 0; f < 4; ++f) o[f] = vzero;

  const float cexp = 0.125f * 1.44269504089f;   // softmax scale * log2(e)
  bf16* pt = &p_lds[wave][0];

  for (int kblk = 0; kblk < SEQ; kblk += 64) {
    if (kblk + 64 < SEQ) {   // prefetch next key/value block
      __builtin_prefetch(kp + (size_t)(kblk + 64 + ll) * HDIM, 0, 3);
      __builtin_prefetch(vp + (size_t)ll * SEQ + kblk + 64, 0, 3);
    }

    // ---- S tiles: s[j] covers keys kblk + j*16 .. +15 ----
    v8f s[4];
    #pragma unroll
    for (int j = 0; j < 4; ++j) s[j] = vzero;
    #pragma unroll
    for (int c2 = 0; c2 < 2; ++c2) {
      #pragma unroll
      for (int j = 0; j < 4; ++j) {
        const bf16* p = kp + (size_t)(kblk + j * 16 + ll) * HDIM + c2 * 32 + kb0;
        BFrag kf;
        kf.h[0] = *(const v8bf*)p; kf.h[1] = *(const v8bf*)(p + 16);
        s[j] = wmma_bf16(qf[c2].v, kf.v, s[j]);
      }
    }

    // ---- online softmax: C-frag row r lives in VGPR r (+8 for lanes 16-31),
    //      identical mapping for S and O accumulators, so m/l index by r. ----
    #pragma unroll
    for (int r = 0; r < 8; ++r) {
      const float v0 = s[0][r], v1 = s[1][r], v2 = s[2][r], v3 = s[3][r];
      float mx = fmaxf(fmaxf(v0, v1), fmaxf(v2, v3));
      mx = fmaxf(mx, __shfl_xor(mx, 1, 32));
      mx = fmaxf(mx, __shfl_xor(mx, 2, 32));
      mx = fmaxf(mx, __shfl_xor(mx, 4, 32));
      mx = fmaxf(mx, __shfl_xor(mx, 8, 32));
      const float mn    = fmaxf(m[r], mx);
      const float alpha = __builtin_amdgcn_exp2f((m[r] - mn) * cexp);
      const float p0    = __builtin_amdgcn_exp2f((v0 - mn) * cexp);
      const float p1    = __builtin_amdgcn_exp2f((v1 - mn) * cexp);
      const float p2    = __builtin_amdgcn_exp2f((v2 - mn) * cexp);
      const float p3    = __builtin_amdgcn_exp2f((v3 - mn) * cexp);
      float rs = (p0 + p1) + (p2 + p3);
      rs += __shfl_xor(rs, 1, 32);
      rs += __shfl_xor(rs, 2, 32);
      rs += __shfl_xor(rs, 4, 32);
      rs += __shfl_xor(rs, 8, 32);
      l[r] = l[r] * alpha + rs;
      m[r] = mn;
      #pragma unroll
      for (int f = 0; f < 4; ++f) o[f][r] = o[f][r] * alpha;
      // stash P in LDS (C-layout -> memory) for A-frag reload
      const int row = r + lg * 8;
      pt[row * PSTRIDE + ll]      = (bf16)p0;
      pt[row * PSTRIDE + 16 + ll] = (bf16)p1;
      pt[row * PSTRIDE + 32 + ll] = (bf16)p2;
      pt[row * PSTRIDE + 48 + ll] = (bf16)p3;
    }
    asm volatile("s_wait_dscnt 0" ::: "memory");   // P stores visible to reloads

    // reload P as two 16x32 A-fragments
    BFrag pf[2];
    #pragma unroll
    for (int c2 = 0; c2 < 2; ++c2) {
      const bf16* pp = pt + ll * PSTRIDE + c2 * 32;
      pf[c2].h[0] = *(const v8bf*)(pp + kb0);
      pf[c2].h[1] = *(const v8bf*)(pp + 16 + kb0);
    }

    // ---- O += P * V  (V transposed: [hd][n], contiguous along keys) ----
    #pragma unroll
    for (int c2 = 0; c2 < 2; ++c2) {
      #pragma unroll
      for (int nt = 0; nt < 4; ++nt) {
        const bf16* vv = vp + (size_t)(nt * 16 + ll) * SEQ + kblk + c2 * 32 + kb0;
        BFrag vf;
        vf.h[0] = *(const v8bf*)vv; vf.h[1] = *(const v8bf*)(vv + 16);
        o[nt] = wmma_bf16(pf[c2].v, vf.v, o[nt]);
      }
    }
  }

  // ---- epilogue: normalize and store to [B][N][H*HD] bf16 ----
  const int bi = bh >> 4, h = bh & 15;
  #pragma unroll
  for (int r = 0; r < 8; ++r) {
    const float inv = 1.0f / l[r];
    const int nrow  = qbase + r + lg * 8;
    const size_t base = ((size_t)(bi * SEQ + nrow)) * EMBED + h * HDIM;
    #pragma unroll
    for (int nt = 0; nt < 4; ++nt)
      attn_ws[base + nt * 16 + ll] = (bf16)(o[nt][r] * inv);
  }
}

// ===========================================================================
// Kernel 3: out = attn @ W_out^T + b_out   (M=8192, K=1024, N=1024, fp32 out)
// Double-buffered LDS staging.
// ===========================================================================
__global__ __launch_bounds__(256)
void mha_out_proj(const bf16* __restrict__ attn, const float* __restrict__ Wout,
                  const float* __restrict__ bias, float* __restrict__ out)
{
  __shared__ alignas(16) bf16 xs [2][128 * LDS_STRIDE];
  __shared__ alignas(16) bf16 wsb[2][128 * LDS_STRIDE];

  const int tid  = threadIdx.x;
  const int lane = tid & 31;
  const int wave = tid >> 5;
  const int ll   = lane & 15;
  const int lg   = lane >> 4;
  const int mbase = blockIdx.x * 128;
  const int nbase = blockIdx.y * 128;
  const int mw = (wave >> 2) * 64;
  const int nw = (wave &  3) * 32;

  const v8f vzero = {0.f,0.f,0.f,0.f,0.f,0.f,0.f,0.f};
  v8f acc[4][2];
  #pragma unroll
  for (int mt = 0; mt < 4; ++mt)
    #pragma unroll
    for (int nt = 0; nt < 2; ++nt) acc[mt][nt] = vzero;

  const int kb0 = lg ? 8  : 0;
  const int kb2 = lg ? 16 : 0;

  auto stage = [&](int buf, int kb) {
    // attn tile is already bf16: straight 16B copies into LDS
    #pragma unroll
    for (int i = 0; i < 2; ++i) {
      const int idx = tid + i * 256;      // 512 v8bf units per tile
      const int row = idx >> 2;           // 4 units per 32-wide row
      const int c8  = (idx & 3) * 8;
      *(v8bf*)&xs[buf][row * LDS_STRIDE + c8] =
          *(const v8bf*)(attn + (size_t)(mbase + row) * EMBED + kb + c8);
    }
    // W_out: fp32 -> bf16 while staging
    #pragma unroll
    for (int i = 0; i < 4; ++i) {
      const int idx = tid + i * 256;
      const int row = idx >> 3;
      const int c4  = (idx & 7) * 4;
      const float4 fw = *(const float4*)(Wout + (size_t)(nbase + row) * EMBED + kb + c4);
      bf16* dw = &wsb[buf][row * LDS_STRIDE + c4];
      dw[0] = (bf16)fw.x; dw[1] = (bf16)fw.y; dw[2] = (bf16)fw.z; dw[3] = (bf16)fw.w;
    }
  };

  stage(0, 0);
  __syncthreads();

  for (int kb = 0; kb < EMBED; kb += 32) {
    const int buf = (kb >> 5) & 1;
    if (kb + 32 < EMBED) stage(buf ^ 1, kb + 32);

    BFrag a[4], b[2];
    #pragma unroll
    for (int mt = 0; mt < 4; ++mt) {
      const bf16* p = &xs[buf][(mw + mt * 16 + ll) * LDS_STRIDE];
      a[mt].h[0] = *(const v8bf*)(p + kb0);
      a[mt].h[1] = *(const v8bf*)(p + 16 + kb0);
    }
    #pragma unroll
    for (int nt = 0; nt < 2; ++nt) {
      const bf16* p = &wsb[buf][(nw + nt * 16 + ll) * LDS_STRIDE + kb0];
      b[nt].h[0] = *(const v8bf*)(p);
      b[nt].h[1] = *(const v8bf*)(p + 16);
    }
    #pragma unroll
    for (int mt = 0; mt < 4; ++mt)
      #pragma unroll
      for (int nt = 0; nt < 2; ++nt)
        acc[mt][nt] = wmma_bf16(a[mt].v, b[nt].v, acc[mt][nt]);
    __syncthreads();
  }

  #pragma unroll
  for (int mt = 0; mt < 4; ++mt) {
    #pragma unroll
    for (int nt = 0; nt < 2; ++nt) {
      const int n = nbase + nw + nt * 16 + ll;
      const float bn = bias[n];
      #pragma unroll
      for (int r = 0; r < 8; ++r) {
        const int mm = mbase + mw + mt * 16 + r + lg * 8;
        out[(size_t)mm * EMBED + n] = acc[mt][nt][r] + bn;
      }
    }
  }
}

// ===========================================================================
extern "C" void kernel_launch(void* const* d_in, const int* in_sizes, int n_in,
                              void* d_out, int out_size, void* d_ws, size_t ws_size,
                              hipStream_t stream)
{
  (void)in_sizes; (void)n_in; (void)out_size; (void)ws_size;
  const float* x     = (const float*)d_in[0];   // [4,2048,1024]
  const float* Wqkv  = (const float*)d_in[1];   // [3072,1024]
  const float* Wout  = (const float*)d_in[2];   // [1024,1024]
  const float* bout  = (const float*)d_in[3];   // [1024]
  float* out = (float*)d_out;                   // [4,2048,1024]

  // workspace: q(16MB) | k(16MB) | v^T(16MB) | attn(16MB), all bf16
  const size_t SLAB = (size_t)BATCH * HEADS * SEQ * HDIM * sizeof(bf16); // 16 MiB
  char* ws = (char*)d_ws;
  bf16* q_ws    = (bf16*)(ws);
  bf16* k_ws    = (bf16*)(ws + SLAB);
  bf16* v_ws    = (bf16*)(ws + 2 * SLAB);
  bf16* attn_ws = (bf16*)(ws + 3 * SLAB);

  dim3 blk(256);
  // 1) QKV projection: M=8192 / 128 = 64, N=3072 / 128 = 24
  mha_qkv_gemm<<<dim3(64, 24), blk, 0, stream>>>(x, Wqkv, q_ws, k_ws, v_ws);
  // 2) Flash attention: B*H * (N/128) = 64 * 16 = 1024 blocks
  mha_flash_attn<<<dim3(1024), blk, 0, stream>>>(q_ws, k_ws, v_ws, attn_ws);
  // 3) Output projection: 64 x (1024/128 = 8)
  mha_out_proj<<<dim3(64, 8), blk, 0, stream>>>(attn_ws, Wout, bout, out);
}
